// CapacityRecoveryDetector_79534204388117
// MI455X (gfx1250) — matrix operand order repaired and checked by reading, not compiled, not run on hardware
//
#include <hip/hip_runtime.h>
#include <hip/hip_bf16.h>

typedef __attribute__((ext_vector_type(16))) __bf16 v16bf;
typedef __attribute__((ext_vector_type(8)))  __bf16 v8bf;
typedef __attribute__((ext_vector_type(8)))  float  v8f;

#define Bdim 16
#define Cdim 256
#define Tdim 4096
#define Kdim 5
#define TT   64          // t-tile per block
#define STP  272         // transposed slab row stride in halves (136 dw == 8 mod 64 banks)
#define KRED 768         // C*3 GEMM reduction length
#define NSTEP 24         // KRED / 32

union AFrag { v16bf v; v8bf h2[2]; };

__device__ __forceinline__ v8f wmma_bf16(v16bf a, v16bf b, v8f c) {
    // D = A(16x32 bf16) * B(32x16 bf16) + C(16x16 f32)
    return __builtin_amdgcn_wmma_f32_16x16x32_bf16(false, a, false, b, (short)0, c, false, false);
}

__device__ __forceinline__ float fast_sigmoid(float z) {
    return __builtin_amdgcn_rcpf(1.0f + __expf(-z));   // v_rcp_f32, v_exp_f32
}

// A-fragment: slab stored transposed S_T[tt][c]; lane's 16 halves are two
// contiguous 8-half (16B) runs -> two ds_load_b128, bank-conflict-free.
__device__ __forceinline__ v16bf load_afrag(const __bf16* S, int tt, int cb, int hiK) {
    const int base = tt * STP + cb + (hiK << 4);
    AFrag af;
    af.h2[0] = *reinterpret_cast<const v8bf*>(S + base);
    af.h2[1] = *reinterpret_cast<const v8bf*>(S + base + 8);
    return af.v;
}

// ---------------------------------------------------------------------------
// Kernel 1: offset/mod conv via WMMA + deformable sampling -> deformed (B,C,T)
// ---------------------------------------------------------------------------
__global__ __launch_bounds__(256) void k1_offset_deform(
    const float* __restrict__ x,        const float* __restrict__ offset_w,
    const float* __restrict__ offset_b, const float* __restrict__ mod_w,
    const float* __restrict__ mod_b,    const float* __restrict__ weight,
    float* __restrict__ deformed)
{
    __shared__ __attribute__((aligned(32))) __bf16 S[66 * STP];      // 35904 B
    __shared__ __attribute__((aligned(32))) __bf16 Bk[NSTEP * 512];  // 24576 B
    __shared__ float OFF[TT][8];                                     //  2048 B
    __shared__ float MODV[TT][8];                                    //  2048 B

    const int tid  = threadIdx.x;
    const int lane = tid & 31;
    const int wv   = tid >> 5;
    const int t0   = blockIdx.x * TT;
    const int b    = blockIdx.y;
    const float* xb0 = x + (size_t)b * Cdim * Tdim;

    // Stage x slab transposed: S[tt][c], tt in [0,66) <-> t in [t0-1, t0+64].
    for (int idx = tid; idx < Cdim * 66; idx += 256) {
        int c = idx / 66, tt = idx - c * 66;
        int t = t0 - 1 + tt;
        float v = (t >= 0 && t < Tdim) ? xb0[c * Tdim + t] : 0.0f;
        S[tt * STP + c] = (__bf16)v;
    }
    // Combined weight (cols: 0..4 offset, 5..9 mod, 10..15 zero) in WMMA
    // B-fragment layout Bk[s][lane][half]; GEMM k-index ordering k = j*256 + c.
    for (int idx = tid; idx < NSTEP * 512; idx += 256) {
        int s = idx >> 9, rem = idx & 511, L = rem >> 4, h = rem & 15;
        int o = L & 15, hiK = L >> 4;
        int kk = h + (hiK << 4);
        int j = s >> 3, c = ((s & 7) << 5) + kk;
        float v = 0.0f;
        if (o < 5)       v = offset_w[o * KRED + c * 3 + j];
        else if (o < 10) v = mod_w[(o - 5) * KRED + c * 3 + j];
        Bk[idx] = (__bf16)v;
    }
    __syncthreads();

    if (wv < 4) {   // waves 0..3: one 16-row m-block each; 64x16x768 GEMM
        const int mb  = wv;
        const int Lm  = lane & 15;
        const int hiK = lane >> 4;
        v8f acc = {};
        for (int s = 0; s < NSTEP; ++s) {
            int j = s >> 3, cb = (s & 7) << 5;
            v16bf a  = load_afrag(S, (mb << 4) + Lm + j, cb, hiK);
            v16bf bb = *reinterpret_cast<const v16bf*>(Bk + (s << 9) + (lane << 4));
            acc = wmma_bf16(a, bb, acc);
        }
        // Scatter D: n<5 -> offsets (+bias); 5<=n<10 -> sigmoid(mod conv + bias)
        const int n = lane & 15;
#pragma unroll
        for (int r = 0; r < 8; ++r) {
            int tl = (mb << 4) + r + (hiK << 3);
            float v = acc[r];
            if (n < 5)       OFF[tl][n]      = v + offset_b[n];
            else if (n < 10) MODV[tl][n - 5] = fast_sigmoid(v + mod_b[n - 5]);
        }
    }
    __syncthreads();

    // Deformable sampling + diagonal einsum. Gathers hit L2 (x fits in 192MB L2).
    const int tl = tid & 63;
    const int t  = t0 + tl;
    float offv[5], modv[5];
#pragma unroll
    for (int k = 0; k < 5; ++k) { offv[k] = OFF[tl][k]; modv[k] = MODV[tl][k]; }
    for (int i = 0; i < 64; ++i) {
        int c = (i << 2) + (tid >> 6);
        const float* xc = xb0 + c * Tdim;
        float acc = 0.0f;
#pragma unroll
        for (int k = 0; k < 5; ++k) {
            float pos = (float)t + (float)(k - 2) + offv[k];
            pos = fminf(fmaxf(pos, 0.0f), (float)(Tdim - 1));
            float pf = floorf(pos), pc = ceilf(pos);
            float wf = pc - pos, wc = pos - pf;      // matches reference exactly
            int pfi = min(max((int)pf, 0), Tdim - 1);
            int pci = min(max((int)pc, 0), Tdim - 1);
            float sval = xc[pfi] * wf + xc[pci] * wc;
            acc += sval * weight[c * (Cdim * Kdim + Kdim) + k] * modv[k];
        }
        deformed[((size_t)b * Cdim + c) * Tdim + t] = acc;   // coalesced (lanes->t)
    }
}

// ---------------------------------------------------------------------------
// Prepack rp1_w -> bf16 WMMA B-fragment layout in d_ws (one shot, 192 KB):
// W2[s][nb][lane][half], 98304 halves total.
// ---------------------------------------------------------------------------
__global__ __launch_bounds__(256) void k_prepack(const float* __restrict__ rp1_w,
                                                 __bf16* __restrict__ W2)
{
    int idx = blockIdx.x * 256 + threadIdx.x;      // 0 .. 98303
    int s = idx >> 12, rem = idx & 4095;
    int nb = rem >> 9, L = (rem >> 4) & 31, h = rem & 15;
    int o = (nb << 4) + (L & 15), hiK = L >> 4;
    int kk = h + (hiK << 4);
    int j = s >> 3, c = ((s & 7) << 5) + kk;
    W2[idx] = (__bf16)rp1_w[o * KRED + c * 3 + j];
}

// ---------------------------------------------------------------------------
// Kernel 2: rp1 conv GEMM (64x128x768, bf16 WMMA) + fused relu/bias/rp2 reduce
// PREPACKED: B slices copied coalesced from W2; else strided self-staging.
// ---------------------------------------------------------------------------
template <bool PREPACKED>
__global__ __launch_bounds__(256) void k2_recover(
    const float* __restrict__ deformed, const float* __restrict__ rp1_w,
    const __bf16* __restrict__ W2,      const float* __restrict__ rp1_b,
    const float* __restrict__ rp2_w,    const float* __restrict__ rp2_b,
    float* __restrict__ recovery)
{
    __shared__ __attribute__((aligned(32))) __bf16 S[66 * STP];   // 35904 B
    __shared__ __attribute__((aligned(32))) __bf16 Bs[8 * 512];   //  8192 B
    __shared__ float Rec[TT];

    const int tid  = threadIdx.x;
    const int lane = tid & 31;
    const int wv   = tid >> 5;
    const int t0   = blockIdx.x * TT;
    const int b    = blockIdx.y;
    const float* db0 = deformed + (size_t)b * Cdim * Tdim;

    for (int idx = tid; idx < Cdim * 66; idx += 256) {
        int c = idx / 66, tt = idx - c * 66;
        int t = t0 - 1 + tt;
        float v = (t >= 0 && t < Tdim) ? db0[c * Tdim + t] : 0.0f;
        S[tt * STP + c] = (__bf16)v;
    }
    if (tid < TT) Rec[tid] = 0.0f;

    const int mb  = wv & 3;      // m-block (16 t's); two waves per m-block
    const int p   = wv >> 2;     // channel half: n-blocks 4p..4p+3
    const int Lm  = lane & 15;
    const int hiK = lane >> 4;

    v8f acc0 = {}, acc1 = {}, acc2 = {}, acc3 = {};

    // fallback staging indices
    const int snb = tid >> 5;
    const int sL  = tid & 31;
    const int so  = (snb << 4) + (sL & 15);
    const int shi = (sL >> 4) & 1;

    for (int s = 0; s < NSTEP; ++s) {
        const int j = s >> 3, cb = (s & 7) << 5;
        __syncthreads();                 // previous step's Bs reads done
        if (PREPACKED) {
            // one coalesced 32B global load + one 32B ds store per thread
            const v16bf* gW = reinterpret_cast<const v16bf*>(W2) + (s << 8);
            *reinterpret_cast<v16bf*>(Bs + (tid << 4)) = gW[tid];
            if (s + 1 < NSTEP) __builtin_prefetch(gW + 256 + tid, 0, 0);
        } else {
            const float* wrow = rp1_w + (size_t)so * KRED + j;
#pragma unroll
            for (int h = 0; h < 16; ++h) {
                int c = cb + h + (shi << 4);
                Bs[(snb << 9) + (sL << 4) + h] = (__bf16)wrow[c * 3];
            }
        }
        __syncthreads();

        v16bf a = load_afrag(S, (mb << 4) + Lm + j, cb, hiK);
        const __bf16* bbase = Bs + ((p << 2) << 9) + (lane << 4);
        v16bf b0 = *reinterpret_cast<const v16bf*>(bbase);
        v16bf b1 = *reinterpret_cast<const v16bf*>(bbase + 512);
        v16bf b2 = *reinterpret_cast<const v16bf*>(bbase + 1024);
        v16bf b3 = *reinterpret_cast<const v16bf*>(bbase + 1536);
        acc0 = wmma_bf16(a, b0, acc0);
        acc1 = wmma_bf16(a, b1, acc1);
        acc2 = wmma_bf16(a, b2, acc2);
        acc3 = wmma_bf16(a, b3, acc3);
    }

    // h = relu(acc + rp1_b); recovery partial = sum_o h*rp2_w -> LDS reduce
    {
        const int n = lane & 15;
        v8f accs[4] = {acc0, acc1, acc2, acc3};
        float contrib[8];
#pragma unroll
        for (int r = 0; r < 8; ++r) contrib[r] = 0.0f;
#pragma unroll
        for (int q = 0; q < 4; ++q) {
            int o = (((p << 2) + q) << 4) + n;
            float bias = rp1_b[o];
            float w2   = rp2_w[o];
#pragma unroll
            for (int r = 0; r < 8; ++r) {
                float hv = fmaxf(accs[q][r] + bias, 0.0f);
                contrib[r] += hv * w2;
            }
        }
#pragma unroll
        for (int r = 0; r < 8; ++r) {
            int m = (mb << 4) + r + (hiK << 3);
            atomicAdd(&Rec[m], contrib[r]);
        }
    }
    __syncthreads();
    if (tid < TT) recovery[(size_t)b * Tdim + t0 + tid] = Rec[tid] + rp2_b[0];
}

// ---------------------------------------------------------------------------
extern "C" void kernel_launch(void* const* d_in, const int* in_sizes, int n_in,
                              void* d_out, int out_size, void* d_ws, size_t ws_size,
                              hipStream_t stream) {
    (void)in_sizes; (void)n_in; (void)out_size;
    const float* x        = (const float*)d_in[0];
    const float* offset_w = (const float*)d_in[1];
    const float* offset_b = (const float*)d_in[2];
    const float* mod_w    = (const float*)d_in[3];
    const float* mod_b    = (const float*)d_in[4];
    const float* weight   = (const float*)d_in[5];
    const float* rp1_w    = (const float*)d_in[6];
    const float* rp1_b    = (const float*)d_in[7];
    const float* rp2_w    = (const float*)d_in[8];
    const float* rp2_b    = (const float*)d_in[9];

    float* out      = (float*)d_out;
    float* deformed = out;                                   // (B,C,T) first
    float* recovery = out + (size_t)Bdim * Cdim * Tdim;      // then (B,T)

    dim3 grid(Tdim / TT, Bdim);
    k1_offset_deform<<<grid, 256, 0, stream>>>(x, offset_w, offset_b, mod_w,
                                               mod_b, weight, deformed);

    const size_t W2_bytes = (size_t)NSTEP * 4096 * sizeof(__bf16);  // 196608 B
    if (ws_size >= W2_bytes) {
        __bf16* W2 = (__bf16*)d_ws;
        k_prepack<<<dim3(NSTEP * 4096 / 256), 256, 0, stream>>>(rp1_w, W2);
        k2_recover<true><<<grid, 256, 0, stream>>>(deformed, rp1_w, W2, rp1_b,
                                                   rp2_w, rp2_b, recovery);
    } else {
        k2_recover<false><<<grid, 256, 0, stream>>>(deformed, rp1_w, nullptr, rp1_b,
                                                    rp2_w, rp2_b, recovery);
    }
}
